// RNNDecoder_88493506167465
// MI455X (gfx1250) — compile-verified, matrix operand order checked by reading, not run on hardware
//
#include <hip/hip_runtime.h>
#include <hip/hip_bf16.h>

#define B_  64
#define N_  64
#define T_  60
#define D_  4
#define H_  64
#define ET_ 4
#define E_  4032   // N*(N-1)

typedef __attribute__((ext_vector_type(16))) _Float16 v16h;
typedef __attribute__((ext_vector_type(8)))  float    v8f;

// K index inside a 32-wide chunk for the 16-bit WMMA A/B operand layout
// (CDNA5 ISA 7.12.2): VGPR v holds packed pair p; lanes 16-31 shift K by 8.
__device__ __forceinline__ int kmap(int lane, int j) {
  int v = j >> 1, p = j & 1;
  return ((v >> 2) << 4) + ((lane >> 4) << 3) + ((v & 3) << 1) + p;
}

// ---------------------------------------------------------------------------
// Edge-message kernel: one wave per (batch b, receiver i).
// agg[b,i,:] = cln_msg[b,i,:] + (1/(ET*D)) * sum_{e: recv(e)=i} sum_t
//              rel_type[b,e,t] * tanh(tanh([h_i,h_j] W1_t + b1_t) W2_t + b2_t)
// All WMMA B-operands served from LDS (112 KB/WG, CDNA5 allows 320 KB).
// ---------------------------------------------------------------------------
__global__ __launch_bounds__(256) void edge_msg_kernel(
    const _Float16* __restrict__ h16,
    const float*    __restrict__ rel_type,
    const _Float16* __restrict__ w1p,   // [t][c(4)][nt(4)][lane(32)][16] f16
    const _Float16* __restrict__ w2p,   // [t][c(2)][nt(4)][lane(32)][16] f16
    const float*    __restrict__ msg_b1,
    const float*    __restrict__ msg_b2,
    const float*    __restrict__ cln,
    float*          __restrict__ agg)
{
  __shared__ _Float16 m1S[8][16][64];              // 16 KB: per-wave act tile
  __shared__ _Float16 w1S[ET_ * 4 * 4 * 32 * 16];  // 64 KB: layer-1 weights
  __shared__ _Float16 w2S[ET_ * 2 * 4 * 32 * 16];  // 32 KB: layer-2 weights

  // cooperative copy: weights -> LDS (amortized over 8 waves x 16 iterations)
  {
    const uint4* s1 = (const uint4*)w1p;
    uint4*       d1 = (uint4*)w1S;
    for (int q = threadIdx.x; q < (ET_ * 4 * 4 * 32 * 16) / 8; q += 256)
      d1[q] = s1[q];
    const uint4* s2 = (const uint4*)w2p;
    uint4*       d2 = (uint4*)w2S;
    for (int q = threadIdx.x; q < (ET_ * 2 * 4 * 32 * 16) / 8; q += 256)
      d2[q] = s2[q];
  }
  __syncthreads();

  const int lane = threadIdx.x & 31;
  const int wv   = threadIdx.x >> 5;
  const int idx  = blockIdx.x * 8 + wv;
  const int b    = idx >> 6;
  const int i    = idx & 63;
  const int m    = lane & 15;           // row of the 16-edge tile this lane feeds

  const _Float16* hrecv = h16 + (size_t)(b * N_ + i) * H_;

  // receiver half of the A operand (rows identical) — hoisted out of tile loop
  v16h aRecv[2];
#pragma unroll
  for (int c = 0; c < 2; ++c)
#pragma unroll
    for (int j = 0; j < 16; ++j)
      aRecv[c][j] = hrecv[c * 32 + kmap(lane, j)];

  float tot[4] = {0.f, 0.f, 0.f, 0.f};

  for (int tile = 0; tile < 4; ++tile) {
    // sender half of the A operand (row-dependent gather)
    int jj  = tile * 16 + m;
    int snd = (jj < 63) ? (jj + (jj >= i ? 1 : 0)) : i;   // pad row -> dummy
    const _Float16* hsend = h16 + (size_t)(b * N_ + snd) * H_;
    v16h aSend[2];
#pragma unroll
    for (int c = 0; c < 2; ++c)
#pragma unroll
      for (int j = 0; j < 16; ++j)
        aSend[c][j] = hsend[c * 32 + kmap(lane, j)];

    v8f tacc[4];
#pragma unroll
    for (int nt = 0; nt < 4; ++nt)
#pragma unroll
      for (int v = 0; v < 8; ++v) tacc[nt][v] = 0.f;

    for (int t = 0; t < ET_; ++t) {
      // ---- layer 1: 16x128 @ 128x64, 4 k-chunks x 4 n-tiles = 16 WMMAs ----
      v8f acc1[4];
#pragma unroll
      for (int nt = 0; nt < 4; ++nt) {
        float bb = msg_b1[t * H_ + nt * 16 + m];
#pragma unroll
        for (int v = 0; v < 8; ++v) acc1[nt][v] = bb;
      }
#pragma unroll
      for (int c = 0; c < 4; ++c) {
        v16h av = (c < 2) ? aRecv[c] : aSend[c - 2];
#pragma unroll
        for (int nt = 0; nt < 4; ++nt) {
          v16h bwl = *(const v16h*)(w1S + ((((t * 4 + c) * 4 + nt) * 32 + lane) << 4));
          acc1[nt] = __builtin_amdgcn_wmma_f32_16x16x32_f16(
              false, av, false, bwl, (short)0, acc1[nt], false, false);
        }
      }
      // tanh, stage through LDS to flip D-layout (lane=col) -> A-layout (lane=row)
#pragma unroll
      for (int nt = 0; nt < 4; ++nt)
#pragma unroll
        for (int v = 0; v < 8; ++v) {
          int row = ((lane >> 4) << 3) + v;
          m1S[wv][row][nt * 16 + m] = (_Float16)tanhf(acc1[nt][v]);
        }
      asm volatile("s_wait_dscnt 0" ::: "memory");

      v16h a2[2];
#pragma unroll
      for (int c = 0; c < 2; ++c)
#pragma unroll
        for (int j = 0; j < 16; ++j)
          a2[c][j] = m1S[wv][m][c * 32 + kmap(lane, j)];

      // ---- layer 2: 16x64 @ 64x64, 2 k-chunks x 4 n-tiles = 8 WMMAs ----
      v8f acc2[4];
#pragma unroll
      for (int nt = 0; nt < 4; ++nt) {
        float bb = msg_b2[t * H_ + nt * 16 + m];
#pragma unroll
        for (int v = 0; v < 8; ++v) acc2[nt][v] = bb;
      }
#pragma unroll
      for (int c = 0; c < 2; ++c)
#pragma unroll
        for (int nt = 0; nt < 4; ++nt) {
          v16h bwl = *(const v16h*)(w2S + ((((t * 2 + c) * 4 + nt) * 32 + lane) << 4));
          acc2[nt] = __builtin_amdgcn_wmma_f32_16x16x32_f16(
              false, a2[c], false, bwl, (short)0, acc2[nt], false, false);
        }
      // rel_type row weighting (pad rows get weight 0)
#pragma unroll
      for (int v = 0; v < 8; ++v) {
        int row = ((lane >> 4) << 3) + v;
        int ejj = tile * 16 + row;
        float w = 0.f;
        if (ejj < 63)
          w = rel_type[(size_t)(b * E_ + i * 63 + ejj) * ET_ + t];
#pragma unroll
        for (int nt = 0; nt < 4; ++nt)
          tacc[nt][v] += w * tanhf(acc2[nt][v]);
      }
      // NOTE: no trailing dscnt wait needed — per-wave LDS ops complete in order
    }

    // reduce over the 16 rows of the tile: 8 regs per lane + cross-half shuffle
#pragma unroll
    for (int nt = 0; nt < 4; ++nt) {
      float r = 0.f;
#pragma unroll
      for (int v = 0; v < 8; ++v) r += tacc[nt][v];
      r += __shfl_xor(r, 16, 32);   // wave32: combine lanes l and l+16
      tot[nt] += r;
    }
  }

  if (lane < 16) {
#pragma unroll
    for (int nt = 0; nt < 4; ++nt) {
      int col = nt * 16 + lane;
      size_t o = (size_t)(b * N_ + i) * H_ + col;
      agg[o] = tot[nt] * 0.0625f + cln[o];   // /(ET*D) = /16
    }
  }
}

// ---------------------------------------------------------------------------
// Node update: one wave per (b,n). GRU cell + 3-layer output MLP.
// ---------------------------------------------------------------------------
__global__ __launch_bounds__(256) void node_update_kernel(
    const float* __restrict__ agg,
    float* __restrict__ hbuf, _Float16* __restrict__ h16,
    float* __restrict__ xbuf,
    const float* __restrict__ in_r_w, const float* __restrict__ in_r_b,
    const float* __restrict__ in_z_w, const float* __restrict__ in_z_b,
    const float* __restrict__ in_n_w, const float* __restrict__ in_n_b,
    const float* __restrict__ hr_w,   const float* __restrict__ hr_b,
    const float* __restrict__ hn_w,   const float* __restrict__ hn_b,
    const float* __restrict__ out1_w, const float* __restrict__ out1_b,
    const float* __restrict__ out2_w, const float* __restrict__ out2_b,
    const float* __restrict__ out3_w, const float* __restrict__ out3_b,
    float* __restrict__ out, int tstep)
{
  __shared__ float aggS[8][64];
  __shared__ float hnS[8][64];
  __shared__ float p1S[8][64];
  __shared__ float p2S[8][64];
  const int lane = threadIdx.x & 31;
  const int wv   = threadIdx.x >> 5;
  const int bn   = blockIdx.x * 8 + wv;
  const size_t base = (size_t)bn * H_;

  aggS[wv][lane]      = agg[base + lane];
  aggS[wv][lane + 32] = agg[base + lane + 32];
  float x0 = xbuf[bn * 4 + 0], x1 = xbuf[bn * 4 + 1];
  float x2 = xbuf[bn * 4 + 2], x3 = xbuf[bn * 4 + 3];
  asm volatile("s_wait_dscnt 0" ::: "memory");

#pragma unroll
  for (int cc = 0; cc < 2; ++cc) {
    int col = lane + cc * 32;
    float hra = hr_b[col], hna = hn_b[col];
    for (int k = 0; k < 64; ++k) {
      float a = aggS[wv][k];
      hra = fmaf(a, hr_w[k * 64 + col], hra);
      hna = fmaf(a, hn_w[k * 64 + col], hna);
    }
    float xr = in_r_b[col] + x0 * in_r_w[col]       + x1 * in_r_w[64 + col]
                           + x2 * in_r_w[128 + col] + x3 * in_r_w[192 + col];
    float xz = in_z_b[col] + x0 * in_z_w[col]       + x1 * in_z_w[64 + col]
                           + x2 * in_z_w[128 + col] + x3 * in_z_w[192 + col];
    float xn = in_n_b[col] + x0 * in_n_w[col]       + x1 * in_n_w[64 + col]
                           + x2 * in_n_w[128 + col] + x3 * in_n_w[192 + col];
    float r  = 1.f / (1.f + __expf(-(xr + hra)));
    float z  = 1.f / (1.f + __expf(-(xz + hra)));
    float nn = tanhf(xn + r * hna);
    float hold = hbuf[base + col];
    float hnew = (1.f - z) * nn + z * hold;
    hbuf[base + col] = hnew;
    h16[base + col]  = (_Float16)hnew;
    hnS[wv][col]     = hnew;
  }
  asm volatile("s_wait_dscnt 0" ::: "memory");

#pragma unroll
  for (int cc = 0; cc < 2; ++cc) {
    int col = lane + cc * 32;
    float a = out1_b[col];
    for (int k = 0; k < 64; ++k) a = fmaf(hnS[wv][k], out1_w[k * 64 + col], a);
    p1S[wv][col] = fmaxf(a, 0.f);
  }
  asm volatile("s_wait_dscnt 0" ::: "memory");

#pragma unroll
  for (int cc = 0; cc < 2; ++cc) {
    int col = lane + cc * 32;
    float a = out2_b[col];
    for (int k = 0; k < 64; ++k) a = fmaf(p1S[wv][k], out2_w[k * 64 + col], a);
    p2S[wv][col] = fmaxf(a, 0.f);
  }
  asm volatile("s_wait_dscnt 0" ::: "memory");

  if (lane < 4) {
    float a = out3_b[lane];
    for (int k = 0; k < 64; ++k) a = fmaf(p2S[wv][k], out3_w[k * 4 + lane], a);
    float xv = (lane == 0) ? x0 : (lane == 1) ? x1 : (lane == 2) ? x2 : x3;
    float pred = xv + a;
    out[((size_t)bn * T_ + (tstep + 1)) * D_ + lane] = pred;
    xbuf[bn * 4 + lane] = pred;
  }
}

// ---------------------------------------------------------------------------
// Clinical MLPs (once): blockIdx.x = batch, blockIdx.y = {hidden, cln_msg}
// ---------------------------------------------------------------------------
__global__ __launch_bounds__(256) void clinical_kernel(
    const float* __restrict__ clinical,
    const float* w1a, const float* b1a, const float* w2a, const float* b2a,
    const float* w3a, const float* b3a,
    const float* w1b, const float* b1b, const float* w2b, const float* b2b,
    const float* w3b, const float* b3b,
    float* __restrict__ hbuf, _Float16* __restrict__ h16,
    float* __restrict__ cln)
{
  const int b = blockIdx.x, which = blockIdx.y, tid = threadIdx.x;
  const float* w1 = which ? w1b : w1a; const float* bb1 = which ? b1b : b1a;
  const float* w2 = which ? w2b : w2a; const float* bb2 = which ? b2b : b2a;
  const float* w3 = which ? w3b : w3a; const float* bb3 = which ? b3b : b3a;
  __shared__ float cs[16], h1[64], h2[128];
  if (tid < 16) cs[tid] = clinical[b * 16 + tid];
  __syncthreads();
  if (tid < 64) {
    float a = bb1[tid];
    for (int k = 0; k < 16; ++k) a = fmaf(cs[k], w1[k * 64 + tid], a);
    h1[tid] = fmaxf(a, 0.f);
  }
  __syncthreads();
  if (tid < 128) {
    float a = bb2[tid];
    for (int k = 0; k < 64; ++k) a = fmaf(h1[k], w2[k * 128 + tid], a);
    h2[tid] = fmaxf(a, 0.f);
  }
  __syncthreads();
  for (int col = tid; col < N_ * H_; col += 256) {
    float a = bb3[col];
    for (int k = 0; k < 128; ++k) a = fmaf(h2[k], w3[k * (N_ * H_) + col], a);
    size_t o = (size_t)b * (N_ * H_) + col;
    if (which == 0) { hbuf[o] = a; h16[o] = (_Float16)a; }
    else            { cln[o] = a; }
  }
}

// ---------------------------------------------------------------------------
// Weight packing into per-lane WMMA B-operand layout + x0 / out[t=0] init
// ---------------------------------------------------------------------------
__global__ __launch_bounds__(256) void prep_kernel(
    const float* __restrict__ msg_w1, const float* __restrict__ msg_w2,
    const float* __restrict__ data,
    _Float16* __restrict__ w1p, _Float16* __restrict__ w2p,
    float* __restrict__ xbuf, float* __restrict__ out)
{
  int gid = blockIdx.x * 256 + threadIdx.x;        // 65536 work items
  if (gid < 32768) {                                // w1p
    int j = gid & 15, lane = (gid >> 4) & 31, nt = (gid >> 9) & 3;
    int c = (gid >> 11) & 3, t = gid >> 13;
    int k = c * 32 + kmap(lane, j);
    int col = nt * 16 + (lane & 15);
    w1p[gid] = (_Float16)msg_w1[(t * 128 + k) * 64 + col];
  } else if (gid < 49152) {                         // w2p
    int g = gid - 32768;
    int j = g & 15, lane = (g >> 4) & 31, nt = (g >> 9) & 3;
    int c = (g >> 11) & 1, t = g >> 12;
    int k = c * 32 + kmap(lane, j);
    int col = nt * 16 + (lane & 15);
    w2p[g] = (_Float16)msg_w2[(t * 64 + k) * 64 + col];
  } else {                                          // x0 + out[:, :, 0, :]
    int g = gid - 49152;                            // bn*4+d over B*N*D
    int bn = g >> 2, d = g & 3;
    float v = data[((size_t)bn * T_ + 0) * D_ + d]; // data[b,n,0,d]
    xbuf[g] = v;
    out[((size_t)bn * T_ + 0) * D_ + d] = v;
  }
}

// ---------------------------------------------------------------------------
extern "C" void kernel_launch(void* const* d_in, const int* in_sizes, int n_in,
                              void* d_out, int out_size, void* d_ws, size_t ws_size,
                              hipStream_t stream) {
  const float* data     = (const float*)d_in[0];
  const float* rel_type = (const float*)d_in[1];
  const float* clinical = (const float*)d_in[4];
  const float* msg_w1   = (const float*)d_in[5];
  const float* msg_b1   = (const float*)d_in[6];
  const float* msg_w2   = (const float*)d_in[7];
  const float* msg_b2   = (const float*)d_in[8];
  const float* clinh_w1 = (const float*)d_in[9];
  const float* clinh_b1 = (const float*)d_in[10];
  const float* clinh_w2 = (const float*)d_in[11];
  const float* clinh_b2 = (const float*)d_in[12];
  const float* clinh_w3 = (const float*)d_in[13];
  const float* clinh_b3 = (const float*)d_in[14];
  const float* clinm_w1 = (const float*)d_in[15];
  const float* clinm_b1 = (const float*)d_in[16];
  const float* clinm_w2 = (const float*)d_in[17];
  const float* clinm_b2 = (const float*)d_in[18];
  const float* clinm_w3 = (const float*)d_in[19];
  const float* clinm_b3 = (const float*)d_in[20];
  const float* in_r_w = (const float*)d_in[21]; const float* in_r_b = (const float*)d_in[22];
  const float* in_z_w = (const float*)d_in[23]; const float* in_z_b = (const float*)d_in[24];
  const float* in_n_w = (const float*)d_in[25]; const float* in_n_b = (const float*)d_in[26];
  const float* hr_w   = (const float*)d_in[27]; const float* hr_b   = (const float*)d_in[28];
  const float* hn_w   = (const float*)d_in[29]; const float* hn_b   = (const float*)d_in[30];
  const float* out1_w = (const float*)d_in[31]; const float* out1_b = (const float*)d_in[32];
  const float* out2_w = (const float*)d_in[33]; const float* out2_b = (const float*)d_in[34];
  const float* out3_w = (const float*)d_in[35]; const float* out3_b = (const float*)d_in[36];

  char* ws = (char*)d_ws;
  float*    agg  = (float*)(ws + 0);                 // 1 MB
  float*    hbuf = (float*)(ws + (1u << 20));        // 1 MB
  float*    cln  = (float*)(ws + (2u << 20));        // 1 MB
  _Float16* h16  = (_Float16*)(ws + (3u << 20));     // 0.5 MB
  float*    xbuf = (float*)(ws + 3670016u);          // 64 KB
  _Float16* w1p  = (_Float16*)(ws + 3735552u);       // 64 KB
  _Float16* w2p  = (_Float16*)(ws + 3801088u);       // 32 KB
  float* out = (float*)d_out;

  clinical_kernel<<<dim3(B_, 2), 256, 0, stream>>>(
      clinical,
      clinh_w1, clinh_b1, clinh_w2, clinh_b2, clinh_w3, clinh_b3,
      clinm_w1, clinm_b1, clinm_w2, clinm_b2, clinm_w3, clinm_b3,
      hbuf, h16, cln);
  prep_kernel<<<256, 256, 0, stream>>>(msg_w1, msg_w2, data, w1p, w2p, xbuf, out);

  for (int t = 0; t < T_ - 1; ++t) {
    edge_msg_kernel<<<(B_ * N_) / 8, 256, 0, stream>>>(
        h16, rel_type, w1p, w2p, msg_b1, msg_b2, cln, agg);
    node_update_kernel<<<(B_ * N_) / 8, 256, 0, stream>>>(
        agg, hbuf, h16, xbuf,
        in_r_w, in_r_b, in_z_w, in_z_b, in_n_w, in_n_b,
        hr_w, hr_b, hn_w, hn_b,
        out1_w, out1_b, out2_w, out2_b, out3_w, out3_b,
        out, t);
  }
}
